// CustomBertSelfAttention_79078937854339
// MI455X (gfx1250) — matrix operand order repaired and checked by reading, not compiled, not run on hardware
//
#include <hip/hip_runtime.h>
#include <stdint.h>

// ---------------- problem constants ----------------
#define BB   2
#define SS   2048
#define DD   1024
#define HH   16
#define DHH  64
#define NPOS (2 * 2048 - 1)   // 4095 rows of dist_emb

// ---------------- WMMA / vector types ----------------
typedef __attribute__((ext_vector_type(16))) __bf16          v16bf;
typedef __attribute__((ext_vector_type(2)))  __bf16          v2bf;
typedef __attribute__((ext_vector_type(8)))  float           v8f;
typedef __attribute__((ext_vector_type(8)))  unsigned int    v8u;
typedef __attribute__((ext_vector_type(4)))  unsigned int    v4u;
typedef __attribute__((ext_vector_type(2)))  unsigned int    v2u;
typedef __attribute__((ext_vector_type(4)))  float           v4f;
typedef __attribute__((ext_vector_type(4)))  int             v4i;

#define WMMA_BF16(A, Bm, C) \
  __builtin_amdgcn_wmma_f32_16x16x32_bf16(false, (A), false, (Bm), (short)0, (C), false, false)

// Async global->LDS copy path (gfx1250 GLOBAL_LOAD_ASYNC_TO_LDS_B128), guarded.
#if defined(__has_builtin)
#if __has_builtin(__builtin_amdgcn_global_load_async_to_lds_b128) && \
    __has_builtin(__builtin_amdgcn_s_wait_asynccnt)
#define USE_ASYNC_COPY 1
#endif
#endif
typedef __attribute__((address_space(1))) v4i* GV4Ptr;
typedef __attribute__((address_space(3))) v4i* LV4Ptr;

// ---- bf16 conversions: native casts -> v_cvt_pk_bf16_f32 ----
__device__ __forceinline__ unsigned short f2bf(float f) {
  return __builtin_bit_cast(unsigned short, (__bf16)f);
}
__device__ __forceinline__ unsigned int f2bf_pk(float a, float b) {
  v2bf r = { (__bf16)a, (__bf16)b };
  return __builtin_bit_cast(unsigned int, r);
}
__device__ __forceinline__ float bf2f(unsigned short h) {
  unsigned int u = ((unsigned int)h) << 16;
  return __builtin_bit_cast(float, u);
}

// Build a 16-bf16 fragment from two 16-byte chunks at p and p+hi_off (ushorts).
// A-fragment: p = row_base + 16*kh, hi_off = 8   (16 contiguous K ushorts)
// B-fragment: p = row_base + 16*kh, hi_off = 8   (16 contiguous K ushorts)
__device__ __forceinline__ v16bf frag_from(const unsigned short* p, int hi_off) {
  v4u lo = *(const v4u*)p;
  v4u hi = *(const v4u*)(p + hi_off);
  v8u u  = __builtin_shufflevector(lo, hi, 0, 1, 2, 3, 4, 5, 6, 7);
  return __builtin_bit_cast(v16bf, u);
}

// =====================================================================
// Kernel 1: GEMM  Out[b,h,s,dh](bf16) = X[b,s,:] @ W[:,n] + bias[n]
// 64x64 output tile / workgroup, 4 waves; X and W^T staged in LDS (bf16).
// =====================================================================
__global__ __launch_bounds__(128) void qkv_gemm_kernel(
    const float* __restrict__ X,      // [B*S, D]
    const float* __restrict__ W,      // [D, D]
    const float* __restrict__ bias,   // [D]
    unsigned short* __restrict__ Out) // [B*H, S, DH] bf16
{
  __shared__ unsigned short Xs[64][40];  // [m][k]  (stride mult of 8 -> 16B align)
  __shared__ unsigned short Wt[64][40];  // [n][k]  (transposed: K contiguous)

  const int t  = threadIdx.x;
  const int ln = t & 31;
  const int w  = t >> 5;
  const int n0 = blockIdx.x * 64;
  const int m0 = blockIdx.y * 64;
  const int M  = ln & 15;
  const int kh = ln >> 4;

  v8f acc[4] = {{}, {}, {}, {}};

  for (int kk = 0; kk < DD; kk += 32) {
    __syncthreads();
    // X tile: 64x32 floats -> bf16 (float4 load, packed uint2 store)
#pragma unroll
    for (int i = 0; i < 4; ++i) {
      int chunk = t + i * 128;             // 512 float4 chunks
      int r = chunk >> 3, c4 = (chunk & 7) * 4;
      v4f x = *(const v4f*)&X[(size_t)(m0 + r) * DD + kk + c4];
      v2u st = { f2bf_pk(x.x, x.y), f2bf_pk(x.z, x.w) };
      *(v2u*)&Xs[r][c4] = st;
    }
    // W tile: transpose during global load (column strips, lane index walks n)
#pragma unroll
    for (int i = 0; i < 4; ++i) {
      int chunk = t + i * 128;             // 512 chunks: n(64) x r4-group(8)
      int n = chunk & 63, r4 = (chunk >> 6) * 4;
      const float* wp = &W[(size_t)(kk + r4) * DD + n0 + n];
      v2u st = { f2bf_pk(wp[0], wp[DD]), f2bf_pk(wp[2 * DD], wp[3 * DD]) };
      *(v2u*)&Wt[n][r4] = st;
    }
    __syncthreads();

    v16bf afrag = frag_from(&Xs[w * 16 + M][16 * kh], 8);
#pragma unroll
    for (int nc = 0; nc < 4; ++nc) {
      v16bf bfrag = frag_from(&Wt[nc * 16 + M][16 * kh], 8);
      acc[nc] = WMMA_BF16(afrag, bfrag, acc[nc]);
    }
  }

  // epilogue: tiles are 64-aligned so (bb, s_base, hh) are block constants.
  const int bb = m0 / SS;                   // batch (tile never straddles)
  const int sB = m0 - bb * SS + w * 16;     // sequence base for this wave
  const int hh = n0 / DHH;                  // head (tile never straddles)
  // Out[((bb*HH + hh)*SS + s)*DHH + dh], s = sB + i, dh = nc*16 + M
  unsigned short* orow =
      Out + (((size_t)(bb * HH + hh)) * SS + sB) * DHH + M;
#pragma unroll
  for (int nc = 0; nc < 4; ++nc) {
    float bv = bias[n0 + nc * 16 + M];
#pragma unroll
    for (int v = 0; v < 8; ++v) {
      int i = v + 8 * kh;                  // C layout row
      orow[(size_t)i * DHH + nc * 16] = f2bf(acc[nc][v] + bv);
    }
  }
}

// =====================================================================
// Kernel 2: fp32 -> bf16 convert (dist_emb)
// =====================================================================
__global__ void cvt_bf16_kernel(const float* __restrict__ in,
                                unsigned short* __restrict__ out, int n) {
  int i = blockIdx.x * blockDim.x + threadIdx.x;
  if (i < n) out[i] = f2bf(in[i]);
}

// =====================================================================
// Kernel 3: flash attention with relative-position bias.
// grid = (S/64, B*H); block = 128 (4 waves); wave w owns rows l0+16w..+15.
// =====================================================================
__global__ __launch_bounds__(128) void flash_attn_kernel(
    const unsigned short* __restrict__ Qb,   // [B*H,S,DH] bf16
    const unsigned short* __restrict__ Kb,
    const unsigned short* __restrict__ Vb,
    const unsigned short* __restrict__ Eb,   // [4095,DH] bf16
    const float* __restrict__ gmask,         // [S,S]
    const float* __restrict__ amask,         // [B,S]
    float* __restrict__ Out)                 // [B,S,D]
{
  __shared__ unsigned short Ks[64][72];      // [r][k]   row-major (K contiguous)
  __shared__ unsigned short Vt[64][72];      // [dh][r]  transposed (K=r contiguous)
  __shared__ unsigned short Es[144][72];     // 128-row dist_emb window (+16 slack)
  __shared__ unsigned short GPs[4][16][104]; // per-wave: G (96 cols), then P (64)

  const int t  = threadIdx.x;
  const int ln = t & 31;
  const int w  = t >> 5;
  const int l0 = blockIdx.x * 64;
  const int bh = blockIdx.y;
  const int bb = bh / HH, hh = bh % HH;
  const int M  = ln & 15;
  const int kh = ln >> 4;

  // Q fragments (resident for the whole r loop): two global b128 loads each
  v16bf aq[2];
  {
    const unsigned short* qrow = Qb + ((size_t)bh * SS + (l0 + 16 * w + M)) * DHH;
#pragma unroll
    for (int kc = 0; kc < 2; ++kc)
      aq[kc] = frag_from(qrow + kc * 32 + 16 * kh, 8);
  }

  v8f   oacc[4] = {{}, {}, {}, {}};
  float m_run[8], l_run[8];
#pragma unroll
  for (int v = 0; v < 8; ++v) { m_run[v] = -1e30f; l_run[v] = 0.0f; }

  for (int r0 = 0; r0 < SS; r0 += 64) {
    __syncthreads();  // protect K/V/E tiles from previous iteration readers

    // ---- cooperative tile loads ----
    const size_t kvbase = ((size_t)bh * SS + r0) * DHH;
    // K tile: contiguous b128 copies (async to LDS when available)
#pragma unroll
    for (int i = 0; i < 4; ++i) {
      int chunk = t + i * 128;                    // 512 chunks of 8 ushorts
      int r = chunk >> 3, c8 = (chunk & 7) * 8;
#ifdef USE_ASYNC_COPY
      __builtin_amdgcn_global_load_async_to_lds_b128(
          (GV4Ptr)&Kb[kvbase + (size_t)r * DHH + c8], (LV4Ptr)&Ks[r][c8], 0, 0);
#else
      *(v4u*)&Ks[r][c8] = *(const v4u*)&Kb[kvbase + (size_t)r * DHH + c8];
#endif
    }
    // V tile: transpose during global load (column strips), b64 LDS stores
#pragma unroll
    for (int i = 0; i < 8; ++i) {
      int chunk = t + i * 128;                    // 1024 chunks: dh(64) x r4(16)
      int dh = chunk & 63, r4 = (chunk >> 6) * 4;
      const unsigned short* vp = &Vb[kvbase + (size_t)r4 * DHH + dh];
      v2u st = { (unsigned int)vp[0]       | ((unsigned int)vp[DHH]     << 16),
                 (unsigned int)vp[2 * DHH] | ((unsigned int)vp[3 * DHH] << 16) };
      *(v2u*)&Vt[dh][r4] = st;
    }
    // E window: contiguous b128 copies (async when available)
    const int base = l0 - r0 + 2047;              // m at (i=0,j=0)
#pragma unroll
    for (int i = 0; i < 9; ++i) {
      int chunk = t + i * 128;                    // 1152 chunks of 8 ushorts
      int r = chunk >> 3, c8 = (chunk & 7) * 8;
      int m = base - 63 + r;
      m = m < 0 ? 0 : (m > NPOS - 1 ? NPOS - 1 : m);
#ifdef USE_ASYNC_COPY
      __builtin_amdgcn_global_load_async_to_lds_b128(
          (GV4Ptr)&Eb[(size_t)m * DHH + c8], (LV4Ptr)&Es[r][c8], 0, 0);
#else
      *(v4u*)&Es[r][c8] = *(const v4u*)&Eb[(size_t)m * DHH + c8];
#endif
    }
#ifdef USE_ASYNC_COPY
    __builtin_amdgcn_s_wait_asynccnt(0);
#endif
    __syncthreads();

    // ---- G = Q @ E_window^T  (96 cols covering offsets [16w, 16w+95]) ----
#pragma unroll
    for (int nc = 0; nc < 6; ++nc) {
      v8f g = {};
#pragma unroll
      for (int kc = 0; kc < 2; ++kc) {
        v16bf bfrag = frag_from(&Es[16 * w + nc * 16 + M][kc * 32 + 16 * kh], 8);
        g = WMMA_BF16(aq[kc], bfrag, g);
      }
#pragma unroll
      for (int v = 0; v < 8; ++v)
        GPs[w][v + 8 * kh][nc * 16 + M] = f2bf(g[v]);
    }
    __syncthreads();

    // ---- S = Q @ K^T ----
    v8f sacc[4] = {{}, {}, {}, {}};
#pragma unroll
    for (int nc = 0; nc < 4; ++nc) {
#pragma unroll
      for (int kc = 0; kc < 2; ++kc) {
        v16bf bfrag = frag_from(&Ks[nc * 16 + M][kc * 32 + 16 * kh], 8);
        sacc[nc] = WMMA_BF16(aq[kc], bfrag, sacc[nc]);
      }
    }

    // ---- rel bias (diagonal gather of G), scale, masks; tile max ----
    float tmax[8];
#pragma unroll
    for (int v = 0; v < 8; ++v) tmax[v] = -1e30f;
#pragma unroll
    for (int nc = 0; nc < 4; ++nc) {
#pragma unroll
      for (int v = 0; v < 8; ++v) {
        int i = v + 8 * kh;                 // row within wave slab
        int j = nc * 16 + M;                // col within r tile
        float g  = bf2f(GPs[w][i][i - j + 63]);
        int l = l0 + 16 * w + i, r = r0 + j;
        float sc = (sacc[nc][v] + g) * 0.125f;
        sc = sc * gmask[(size_t)l * SS + r] + amask[bb * SS + r];
        sacc[nc][v] = sc;
        tmax[v] = fmaxf(tmax[v], sc);
      }
    }
#pragma unroll
    for (int v = 0; v < 8; ++v)
      for (int s = 8; s >= 1; s >>= 1)
        tmax[v] = fmaxf(tmax[v], __shfl_xor(tmax[v], s, 32));

    // ---- online softmax update, write P (bf16) to LDS ----
    float alpha[8], rsum[8];
#pragma unroll
    for (int v = 0; v < 8; ++v) {
      float mn = fmaxf(m_run[v], tmax[v]);
      alpha[v] = __expf(m_run[v] - mn);
      m_run[v] = mn;
      rsum[v]  = 0.0f;
    }
#pragma unroll
    for (int nc = 0; nc < 4; ++nc) {
#pragma unroll
      for (int v = 0; v < 8; ++v) {
        float p = __expf(sacc[nc][v] - m_run[v]);
        rsum[v] += p;
        GPs[w][v + 8 * kh][nc * 16 + M] = f2bf(p);   // reuse G buffer for P
      }
    }
#pragma unroll
    for (int v = 0; v < 8; ++v) {
      for (int s = 8; s >= 1; s >>= 1)
        rsum[v] += __shfl_xor(rsum[v], s, 32);
      l_run[v] = l_run[v] * alpha[v] + rsum[v];
    }
#pragma unroll
    for (int nc = 0; nc < 4; ++nc)
#pragma unroll
      for (int v = 0; v < 8; ++v) oacc[nc][v] *= alpha[v];
    __syncthreads();

    // ---- O += P @ V ----
#pragma unroll
    for (int kc = 0; kc < 2; ++kc) {
      v16bf ap = frag_from(&GPs[w][M][kc * 32 + 16 * kh], 8);
#pragma unroll
      for (int nc = 0; nc < 4; ++nc) {
        v16bf bfrag = frag_from(&Vt[nc * 16 + M][kc * 32 + 16 * kh], 8);
        oacc[nc] = WMMA_BF16(ap, bfrag, oacc[nc]);
      }
    }
  }

  // ---- normalize and store ctx[b, l, h*64+dh] ----
  float* orow = Out + ((size_t)bb * SS + (l0 + 16 * w)) * DD + hh * DHH + M;
#pragma unroll
  for (int nc = 0; nc < 4; ++nc) {
#pragma unroll
    for (int v = 0; v < 8; ++v) {
      int i = v + 8 * kh;
      orow[(size_t)i * DD + nc * 16] = oacc[nc][v] / l_run[v];
    }
  }
}

// =====================================================================
extern "C" void kernel_launch(void* const* d_in, const int* in_sizes, int n_in,
                              void* d_out, int out_size, void* d_ws, size_t ws_size,
                              hipStream_t stream) {
  (void)in_sizes; (void)n_in; (void)out_size; (void)ws_size;

  const float* hs = (const float*)d_in[0];
  const float* am = (const float*)d_in[1];
  const float* gm = (const float*)d_in[2];
  const float* Wq = (const float*)d_in[3];
  const float* bq = (const float*)d_in[4];
  const float* Wk = (const float*)d_in[5];
  const float* bk = (const float*)d_in[6];
  const float* Wv = (const float*)d_in[7];
  const float* bv = (const float*)d_in[8];
  const float* de = (const float*)d_in[9];
  float* out = (float*)d_out;

  // workspace carve: Q,K,V bf16 [B*H,S,DH] + dist_emb bf16  (~25.7 MB)
  const size_t qkvN = (size_t)BB * HH * SS * DHH;
  unsigned short* Qb = (unsigned short*)d_ws;
  unsigned short* Kb = Qb + qkvN;
  unsigned short* Vb = Kb + qkvN;
  unsigned short* Eb = Vb + qkvN;

  dim3 gq(DD / 64, (BB * SS) / 64);
  qkv_gemm_kernel<<<gq, 128, 0, stream>>>(hs, Wq, bq, Qb);
  qkv_gemm_kernel<<<gq, 128, 0, stream>>>(hs, Wk, bk, Kb);
  qkv_gemm_kernel<<<gq, 128, 0, stream>>>(hs, Wv, bv, Vb);

  const int ne = NPOS * DHH;
  cvt_bf16_kernel<<<(ne + 255) / 256, 256, 0, stream>>>(de, Eb, ne);

  flash_attn_kernel<<<dim3(SS / 64, BB * HH), 128, 0, stream>>>(
      Qb, Kb, Vb, Eb, gm, am, out);
}
